// Point_CAE_DGCNN_FCOnly_36764920054479
// MI455X (gfx1250) — matrix operand order repaired and checked by reading, not compiled, not run on hardware
//
#include <hip/hip_runtime.h>
#include <hip/hip_bf16.h>

// ---------------------------------------------------------------------------
// Point_CAE DGCNN encoder + FC decoder + chamfer loss for gfx1250 (MI455X).
//
// EdgeConv factorization:  [x_j - x_i, x_i] @ W = x_j@W_top + x_i@(W_bot-W_top)
// -> one GEMM per layer ([B*N, Cin] x [Cin, 2Cout]) + gather-max (LeakyReLU is
// monotone so the activation commutes with the max).
//
// All GEMM operands are pre-converted to bf16 in fragment-friendly layouts:
//   A: row-major [M, K] bf16 (K padded to mult. of 32)
//   B: transposed [N, K] bf16
// so each lane's WMMA fragment is two contiguous global_load_b128 per matrix
// -- no LDS staging, no per-element guards, inner loop = loads + v_wmma.
// ---------------------------------------------------------------------------

typedef __attribute__((ext_vector_type(16))) __bf16 v16bf;
typedef __attribute__((ext_vector_type(8)))  __bf16 v8bf;
typedef __attribute__((ext_vector_type(8)))  float  v8f;

#define NPTS 2048
#define KNN  20

__device__ __forceinline__ float leaky02(float x) { return x > 0.f ? x : 0.2f * x; }

__device__ __forceinline__ v16bf cat16(v8bf lo, v8bf hi) {
    return __builtin_shufflevector(lo, hi, 0, 1, 2, 3, 4, 5, 6, 7,
                                   8, 9, 10, 11, 12, 13, 14, 15);
}

// ---------------- row squared-norms: sq[i] = sum_c x[i,c]^2 ----------------
__global__ void rownorm_kernel(const float* __restrict__ x, float* __restrict__ sq,
                               int C, int total) {
    int i = blockIdx.x * blockDim.x + threadIdx.x;
    if (i >= total) return;
    const float* p = x + (size_t)i * C;
    float s = 0.f;
    for (int c = 0; c < C; ++c) s += p[c] * p[c];
    sq[i] = s;
}

// ---------------- fused distance-row + top-20 selection --------------------
__global__ void knn_kernel(const float* __restrict__ x, const float* __restrict__ sq,
                           int* __restrict__ idxout, int C) {
    const int bn = blockIdx.x;
    const int b  = bn >> 11;            // NPTS == 2048
    const int n  = bn & (NPTS - 1);
    __shared__ float dist[NPTS];
    __shared__ float xi[128];
    __shared__ float rv[256];
    __shared__ int   ri[256];
    const int tid = threadIdx.x;
    const float* xb = x + (size_t)b * NPTS * C;
    if (tid < C) xi[tid] = xb[(size_t)n * C + tid];
    __syncthreads();
    const float sqi = sq[bn];
    for (int j = tid; j < NPTS; j += 256) {
        const float* xj = xb + (size_t)j * C;
        float dot = 0.f;
        for (int c = 0; c < C; ++c) dot += xi[c] * xj[c];
        dist[j] = sqi + sq[(b << 11) + j] - 2.f * dot;
    }
    __syncthreads();
    for (int t = 0; t < KNN; ++t) {
        float bv = 3.4e38f; int bi = 0;
        for (int j = tid; j < NPTS; j += 256) {
            float d = dist[j];
            if (d < bv) { bv = d; bi = j; }
        }
        rv[tid] = bv; ri[tid] = bi;
        __syncthreads();
        for (int s = 128; s > 0; s >>= 1) {
            if (tid < s && rv[tid + s] < rv[tid]) { rv[tid] = rv[tid + s]; ri[tid] = ri[tid + s]; }
            __syncthreads();
        }
        if (tid == 0) { idxout[(size_t)bn * KNN + t] = ri[0]; dist[ri[0]] = 3.4e38f; }
        __syncthreads();
    }
}

// -------- f32 [rows,C] -> bf16 [rows,Cp], zero-padded in K ----------------
__global__ void f32_to_bf16_pad(const float* __restrict__ in, __bf16* __restrict__ out,
                                int C, int Cp, int rows) {
    int i = blockIdx.x * blockDim.x + threadIdx.x;
    if (i >= rows * Cp) return;
    int r = i / Cp, c = i % Cp;
    out[i] = (__bf16)(c < C ? in[(size_t)r * C + c] : 0.f);
}

// ------- Wcat^T bf16 [2Cout, Cp]: rows 0..Co-1 = W_top^T, rest = (W_bot-W_top)^T
__global__ void build_wcat_t(const float* __restrict__ w, __bf16* __restrict__ wt,
                             int Cin, int Cp, int Cout) {
    int i = blockIdx.x * blockDim.x + threadIdx.x;
    if (i >= 2 * Cout * Cp) return;
    int d = i / Cp, c = i % Cp;
    float v = 0.f;
    if (c < Cin) {
        if (d < Cout) v = w[(size_t)c * Cout + d];
        else {
            int dd = d - Cout;
            v = w[(size_t)(Cin + c) * Cout + dd] - w[(size_t)c * Cout + dd];
        }
    }
    wt[i] = (__bf16)v;
}

// ------------- f32 [K,N] -> bf16 transposed [N,K] --------------------------
__global__ void transpose_to_bf16(const float* __restrict__ in, __bf16* __restrict__ out,
                                  int K, int N) {
    int i = blockIdx.x * blockDim.x + threadIdx.x;
    if (i >= K * N) return;
    int n = i / K, k = i % K;
    out[i] = (__bf16)in[(size_t)k * N + n];
}

// ---------------- direct-fragment bf16 WMMA GEMM ---------------------------
// A:[M,K] bf16 row-major, Bt:[N,K] bf16 (i.e. B transposed), K mult of 32,
// N mult of 32. Out-of-tile A rows (M..31) must be readable (padded buffers);
// stores are guarded by r < M. 128 threads = 4 waves = 2x2 16x16 tiles.
__global__ void gemm_wmma_direct(const __bf16* __restrict__ A, const __bf16* __restrict__ Bt,
                                 const float* __restrict__ bias,
                                 float* __restrict__ Cf, __bf16* __restrict__ Cb,
                                 int M, int N, int K, int act) {
    const int tid  = threadIdx.x;
    const int lane = tid & 31, wave = tid >> 5;
    const int wm = wave >> 1, wn = wave & 1;
    const int hlf = lane >> 4, l16 = lane & 15;
    const int bm = blockIdx.x * 32, bn = blockIdx.y * 32;
    const int col = bn + wn * 16 + l16;                 // B column == D column
    const __bf16* ap = A  + (size_t)(bm + wm * 16 + l16) * K;
    const __bf16* bp = Bt + (size_t)col * K;
    v8f acc = {0.f, 0.f, 0.f, 0.f, 0.f, 0.f, 0.f, 0.f};
#pragma unroll 2
    for (int k0 = 0; k0 < K; k0 += 32) {
        v8bf a0 = *(const v8bf*)(ap + k0 + hlf * 8);
        v8bf a1 = *(const v8bf*)(ap + k0 + 16 + hlf * 8);
        v8bf b0 = *(const v8bf*)(bp + k0 + hlf * 8);
        v8bf b1 = *(const v8bf*)(bp + k0 + 16 + hlf * 8);
        acc = __builtin_amdgcn_wmma_f32_16x16x32_bf16(false, cat16(a0, a1),
                                                      false, cat16(b0, b1),
                                                      (short)0, acc, false, false);
    }
#pragma unroll
    for (int v = 0; v < 8; ++v) {
        int r = bm + wm * 16 + hlf * 8 + v;
        if (r < M) {
            float x = acc[v] + (bias ? bias[col] : 0.f);
            if (act == 1) x = fmaxf(x, 0.f);
            else if (act == 2) x = leaky02(x);
            if (Cf) Cf[(size_t)r * N + col] = x;
            if (Cb) Cb[(size_t)r * N + col] = (__bf16)x;
        }
    }
}

// ------- gather-max: out[b,n,c] = leaky(max_k P[idx[k]][c] + Q[n][c] + b) --
__global__ void edge_gather_max(const float* __restrict__ pq, const int* __restrict__ idx,
                                const float* __restrict__ bias,
                                float* __restrict__ outf, __bf16* __restrict__ outb,
                                int Co) {
    const int bn = blockIdx.x;
    const int b  = bn >> 11;
    __shared__ int nid[KNN];
    if (threadIdx.x < KNN) nid[threadIdx.x] = idx[(size_t)bn * KNN + threadIdx.x];
    __syncthreads();
    const int W2 = 2 * Co;
    const size_t browbase = (size_t)b * NPTS;
    for (int c = threadIdx.x; c < Co; c += blockDim.x) {
        float m = -3.4e38f;
#pragma unroll 4
        for (int k = 0; k < KNN; ++k)
            m = fmaxf(m, pq[(browbase + nid[k]) * (size_t)W2 + c]);
        float q = pq[(size_t)bn * W2 + Co + c];
        float h = leaky02(m + q + bias[c]);
        if (outf) outf[(size_t)bn * Co + c] = h;
        outb[(size_t)bn * Co + c] = (__bf16)h;
    }
}

// ------ fused w5 GEMM + max over N: feat[b,:] = leaky(max_n cat@w5 + b5) ---
// A = cat(x1b..x4b) bf16, read per 32-wide K tile (tiles never straddle a
// buffer boundary). B = w5^T bf16 [1024, 512]. One block per (32 cols, b).
__global__ void feat_w5_max(const __bf16* __restrict__ x1b, const __bf16* __restrict__ x2b,
                            const __bf16* __restrict__ x3b, const __bf16* __restrict__ x4b,
                            const __bf16* __restrict__ w5t, const float* __restrict__ b5,
                            __bf16* __restrict__ featb) {
    __shared__ float red[128];
    const int tid  = threadIdx.x;
    const int lane = tid & 31, wave = tid >> 5;
    const int wm = wave >> 1, wn = wave & 1;
    const int hlf = lane >> 4, l16 = lane & 15;
    const int b  = blockIdx.y;
    const int bn = blockIdx.x * 32;
    const int col = bn + wn * 16 + l16;
    const __bf16* bp = w5t + (size_t)col * 512;
    float runmax[8];
#pragma unroll
    for (int v = 0; v < 8; ++v) runmax[v] = -3.4e38f;

    for (int rt = 0; rt < NPTS / 32; ++rt) {
        const int n = rt * 32 + wm * 16 + l16;          // A fragment row
        v8f acc = {0.f, 0.f, 0.f, 0.f, 0.f, 0.f, 0.f, 0.f};
#pragma unroll 4
        for (int k0 = 0; k0 < 512; k0 += 32) {
            const __bf16* src; int cb, cs;
            if (k0 < 64)       { src = x1b; cb = k0;       cs = 64;  }
            else if (k0 < 128) { src = x2b; cb = k0 - 64;  cs = 64;  }
            else if (k0 < 256) { src = x3b; cb = k0 - 128; cs = 128; }
            else               { src = x4b; cb = k0 - 256; cs = 256; }
            const __bf16* ap = src + ((size_t)b * NPTS + n) * cs + cb;
            v8bf a0 = *(const v8bf*)(ap + hlf * 8);
            v8bf a1 = *(const v8bf*)(ap + 16 + hlf * 8);
            v8bf b0 = *(const v8bf*)(bp + k0 + hlf * 8);
            v8bf b1 = *(const v8bf*)(bp + k0 + 16 + hlf * 8);
            acc = __builtin_amdgcn_wmma_f32_16x16x32_bf16(false, cat16(a0, a1),
                                                          false, cat16(b0, b1),
                                                          (short)0, acc, false, false);
        }
#pragma unroll
        for (int v = 0; v < 8; ++v) runmax[v] = fmaxf(runmax[v], acc[v]);
    }
    float tm = -3.4e38f;
#pragma unroll
    for (int v = 0; v < 8; ++v) tm = fmaxf(tm, runmax[v]);
    red[tid] = tm;
    __syncthreads();
    if (tid < 32) {
        int wn_ = tid >> 4, cl = tid & 15;
        float m = -3.4e38f;
        for (int wm_ = 0; wm_ < 2; ++wm_)
            for (int hf = 0; hf < 2; ++hf)
                m = fmaxf(m, red[(wm_ * 2 + wn_) * 32 + hf * 16 + cl]);
        int c = bn + tid;
        featb[(size_t)b * 1024 + c] = (__bf16)leaky02(m + b5[c]);
    }
}

// ---------------- chamfer: per a-point min squared distance to B -----------
__global__ void chamfer_min(const float* __restrict__ A, const float* __restrict__ B,
                            float* __restrict__ omin, int Bt, int M, int Nn) {
    int i = blockIdx.x * blockDim.x + threadIdx.x;
    if (i >= Bt * M) return;
    int b = i / M, m = i % M;
    const float* ap = A + ((size_t)b * M + m) * 3;
    float ax = ap[0], ay = ap[1], az = ap[2];
    const float* bp = B + (size_t)b * Nn * 3;
    float best = 3.4e38f;
    for (int n = 0; n < Nn; ++n) {
        float dx = ax - bp[n * 3 + 0];
        float dy = ay - bp[n * 3 + 1];
        float dz = az - bp[n * 3 + 2];
        best = fminf(best, dx * dx + dy * dy + dz * dz);
    }
    omin[i] = best;
}

__global__ void final_loss(const float* __restrict__ a, int na,
                           const float* __restrict__ bb, int nb,
                           float* __restrict__ out) {
    __shared__ float sh[256];
    int tid = threadIdx.x;
    float s = 0.f;
    for (int i = tid; i < na; i += 256) s += a[i];
    sh[tid] = s; __syncthreads();
    for (int st = 128; st > 0; st >>= 1) { if (tid < st) sh[tid] += sh[tid + st]; __syncthreads(); }
    float sa = sh[0];
    __syncthreads();
    s = 0.f;
    for (int i = tid; i < nb; i += 256) s += bb[i];
    sh[tid] = s; __syncthreads();
    for (int st = 128; st > 0; st >>= 1) { if (tid < st) sh[tid] += sh[tid + st]; __syncthreads(); }
    if (tid == 0) out[0] = sa / (float)na + sh[0] / (float)nb;
}

// ---------------------------------------------------------------------------
extern "C" void kernel_launch(void* const* d_in, const int* in_sizes, int n_in,
                              void* d_out, int out_size, void* d_ws, size_t ws_size,
                              hipStream_t stream) {
    const float* xin0 = (const float*)d_in[0];
    const float* pts  = (const float*)d_in[1];
    const float* W[4]  = {(const float*)d_in[2], (const float*)d_in[4],
                          (const float*)d_in[6], (const float*)d_in[8]};
    const float* Bv[4] = {(const float*)d_in[3], (const float*)d_in[5],
                          (const float*)d_in[7], (const float*)d_in[9]};
    const float* w5  = (const float*)d_in[10]; const float* b5  = (const float*)d_in[11];
    const float* fw1 = (const float*)d_in[12]; const float* fb1 = (const float*)d_in[13];
    const float* fw2 = (const float*)d_in[14]; const float* fb2 = (const float*)d_in[15];
    const float* fw3 = (const float*)d_in[16]; const float* fb3 = (const float*)d_in[17];
    (void)n_in; (void)out_size; (void)ws_size;

    const int Bt = in_sizes[0] / (NPTS * 3);     // 16
    const int M  = Bt * NPTS;                    // 32768
    const int Mp = 32;                           // padded row count for FC GEMMs

    char* ws = (char*)d_ws;
    size_t off = 0;
    auto carve = [&](size_t bytes) -> void* {
        void* p = ws + off;
        off = (off + bytes + 255) & ~(size_t)255;
        return p;
    };
    // f32 scratch
    float* sq     = (float*)carve((size_t)M * 4);
    int*   idx    = (int*)  carve((size_t)M * KNN * 4);
    float* x1     = (float*)carve((size_t)M * 64  * 4);   // f32 copies feed kNN only
    float* x2     = (float*)carve((size_t)M * 64  * 4);
    float* x3     = (float*)carve((size_t)M * 128 * 4);
    float* pq     = (float*)carve((size_t)M * 512 * 4);   // reused per layer
    float* coarse = (float*)carve((size_t)Mp * 3072 * 4);
    float* minA   = (float*)carve((size_t)Bt * 1024 * 4);
    float* minB   = (float*)carve((size_t)Bt * NPTS * 4);
    // bf16 GEMM operands
    __bf16* x0b   = (__bf16*)carve((size_t)M * 32  * 2);  // corrupted pts, K 3->32
    __bf16* x1b   = (__bf16*)carve((size_t)M * 64  * 2);
    __bf16* x2b   = (__bf16*)carve((size_t)M * 64  * 2);
    __bf16* x3b   = (__bf16*)carve((size_t)M * 128 * 2);
    __bf16* x4b   = (__bf16*)carve((size_t)M * 256 * 2);
    __bf16* wcatt = (__bf16*)carve((size_t)512 * 256 * 2);
    __bf16* w5t   = (__bf16*)carve((size_t)1024 * 512 * 2);
    __bf16* fw1t  = (__bf16*)carve((size_t)1024 * 1024 * 2);
    __bf16* fw2t  = (__bf16*)carve((size_t)1024 * 1024 * 2);
    __bf16* fw3t  = (__bf16*)carve((size_t)3072 * 1024 * 2);
    __bf16* featb = (__bf16*)carve((size_t)Mp * 1024 * 2);
    __bf16* fc1b  = (__bf16*)carve((size_t)Mp * 1024 * 2);
    __bf16* fc2b  = (__bf16*)carve((size_t)Mp * 1024 * 2);

    // one-off operand conversions
    f32_to_bf16_pad<<<(M * 32 + 255) / 256, 256, 0, stream>>>(xin0, x0b, 3, 32, M);
    transpose_to_bf16<<<(512 * 1024 + 255) / 256, 256, 0, stream>>>(w5, w5t, 512, 1024);
    transpose_to_bf16<<<(1024 * 1024 + 255) / 256, 256, 0, stream>>>(fw1, fw1t, 1024, 1024);
    transpose_to_bf16<<<(1024 * 1024 + 255) / 256, 256, 0, stream>>>(fw2, fw2t, 1024, 1024);
    transpose_to_bf16<<<(1024 * 3072 + 255) / 256, 256, 0, stream>>>(fw3, fw3t, 1024, 3072);

    const int Cin[4]  = {3, 64, 64, 128};     // true input channels (kNN)
    const int Kp[4]   = {32, 64, 64, 128};    // padded GEMM K
    const int Cout[4] = {64, 64, 128, 256};
    const float*  lin [4] = {xin0, x1, x2, x3};         // f32 for kNN
    const __bf16* linb[4] = {x0b, x1b, x2b, x3b};       // bf16 GEMM A
    float*  loutf[4] = {x1, x2, x3, nullptr};           // layer-4 f32 unused
    __bf16* loutb[4] = {x1b, x2b, x3b, x4b};

    for (int l = 0; l < 4; ++l) {
        const int C = Cin[l], K = Kp[l], Co = Cout[l];
        rownorm_kernel<<<(M + 255) / 256, 256, 0, stream>>>(lin[l], sq, C, M);
        knn_kernel<<<M, 256, 0, stream>>>(lin[l], sq, idx, C);
        build_wcat_t<<<(2 * Co * K + 255) / 256, 256, 0, stream>>>(W[l], wcatt, C, K, Co);
        dim3 g(M / 32, (2 * Co) / 32);
        gemm_wmma_direct<<<g, 128, 0, stream>>>(linb[l], wcatt, nullptr, pq, nullptr,
                                                M, 2 * Co, K, 0);
        edge_gather_max<<<M, 128, 0, stream>>>(pq, idx, Bv[l], loutf[l], loutb[l], Co);
    }

    feat_w5_max<<<dim3(1024 / 32, Bt), 128, 0, stream>>>(x1b, x2b, x3b, x4b, w5t, b5, featb);

    {
        dim3 g(1, 1024 / 32);
        gemm_wmma_direct<<<g, 128, 0, stream>>>(featb, fw1t, fb1, nullptr, fc1b,
                                                Bt, 1024, 1024, 1);
        gemm_wmma_direct<<<g, 128, 0, stream>>>(fc1b, fw2t, fb2, nullptr, fc2b,
                                                Bt, 1024, 1024, 1);
        dim3 g3(1, 3072 / 32);
        gemm_wmma_direct<<<g3, 128, 0, stream>>>(fc2b, fw3t, fb3, coarse, nullptr,
                                                 Bt, 3072, 1024, 0);
    }

    chamfer_min<<<(Bt * 1024 + 255) / 256, 256, 0, stream>>>(coarse, pts, minA, Bt, 1024, NPTS);
    chamfer_min<<<(Bt * NPTS + 255) / 256, 256, 0, stream>>>(pts, coarse, minB, Bt, NPTS, 1024);
    final_loss<<<1, 256, 0, stream>>>(minA, Bt * 1024, minB, Bt * NPTS, (float*)d_out);
}
